// Multihead_Attention_29180007809153
// MI455X (gfx1250) — matrix operand order repaired and checked by reading, not compiled, not run on hardware
//
#include <hip/hip_runtime.h>
#include <hip/hip_bf16.h>
#include <math.h>

typedef _Float16 f16;
typedef __attribute__((ext_vector_type(16))) _Float16 v16h;
typedef __attribute__((ext_vector_type(8)))  _Float16 v8h;
typedef __attribute__((ext_vector_type(8)))  float    v8f;

#define DM   768
#define HD   96
#define NH   8
#define SEQL 4096
#define BATCH 2

static __device__ __forceinline__ v16h join16(v8h a, v8h b) {
    v16h r;
#pragma unroll
    for (int i = 0; i < 8; ++i) { r[i] = a[i]; r[i + 8] = b[i]; }
    return r;
}

// ---------------------------------------------------------------------------
// Stage 1: C[M x 768] = A[M x 768] @ W[768 x 768], fp32 in, f16 out.
// mode 0: Q  -> [B,H,L,96] scaled by 1/sqrt(96)
// mode 1: K  -> [B,H,L,96]
// mode 2: Vt -> [B,H,96,L]
// 128 threads = 4 waves; block tile 64x64; wave w: rows w*16..w*16+15.
// ---------------------------------------------------------------------------
__global__ __launch_bounds__(128) void proj_kernel(const float* __restrict__ X,
                                                   const float* __restrict__ W,
                                                   f16* __restrict__ out,
                                                   int mode, float scale) {
    __shared__ f16 As[64 * 40];   // [m][k], stride 40 halves (16B-aligned chunks)
    __shared__ f16 Wt[64 * 40];   // [n][k] transposed W tile

    const int tid  = threadIdx.x;
    const int wv   = tid >> 5;
    const int lane = tid & 31;
    const int lo   = lane & 15;
    const int hi   = lane >> 4;
    const int nn   = blockIdx.x * 64;
    const int mm   = blockIdx.y * 64;

    v8f acc[4] = {};

    for (int kk = 0; kk < DM; kk += 32) {
        __syncthreads();
        {   // A tile: 64 rows x 32 k, fp32 -> f16
            int m  = tid >> 1;
            int kh = (tid & 1) * 16;
            const float* src = X + (size_t)(mm + m) * DM + kk + kh;
            f16* dst = As + m * 40 + kh;
#pragma unroll
            for (int i = 0; i < 16; ++i) dst[i] = (f16)src[i];
        }
        {   // W tile transposed: Wt[n][k] = W[kk+k][nn+n]
            int k  = tid >> 2;
            int n0 = (tid & 3) * 16;
            const float* src = W + (size_t)(kk + k) * DM + nn + n0;
#pragma unroll
            for (int i = 0; i < 16; ++i) Wt[(n0 + i) * 40 + k] = (f16)src[i];
        }
        __syncthreads();

        // A fragment: row = wv*16+lo; halves 0..7 -> k = 8*hi.., 8..15 -> k = 16+8*hi..
        const f16* ap = As + (wv * 16 + lo) * 40 + 8 * hi;
        v16h a = join16(*(const v8h*)ap, *(const v8h*)(ap + 16));

#pragma unroll
        for (int f = 0; f < 4; ++f) {
            // B fragment: col = f*16+lo; halves 0..7 -> k = 8*hi.., 8..15 -> k = 16+8*hi..
            const f16* bp = Wt + (f * 16 + lo) * 40 + 8 * hi;
            v16h b = join16(*(const v8h*)bp, *(const v8h*)(bp + 16));
            acc[f] = __builtin_amdgcn_wmma_f32_16x16x32_f16(false, a, false, b,
                                                            (short)0, acc[f], false, false);
        }
    }

    // Epilogue: C element at (row = mm+wv*16+v+8*hi, col = nn+f*16+lo)
#pragma unroll
    for (int f = 0; f < 4; ++f) {
        int c = nn + f * 16 + lo;
        int h = c / HD, d = c % HD;
#pragma unroll
        for (int v = 0; v < 8; ++v) {
            int r = mm + wv * 16 + v + 8 * hi;
            int b = r / SEQL, l = r % SEQL;
            float val = acc[f][v] * scale;
            size_t idx = (mode == 2)
                ? (((size_t)(b * NH + h) * HD + d) * SEQL + l)
                : (((size_t)(b * NH + h) * SEQL + l) * HD + d);
            out[idx] = (f16)val;
        }
    }
}

// ---------------------------------------------------------------------------
// Stage 2: flash attention. 128 threads = 4 waves; block = 64 q rows of one
// (b,h); wave owns 16 q rows. K tile = 64 columns -> 24 WMMA per iteration.
// Wave-private P scratch in LDS; same-wave DS ops are in-order per the ISA,
// so no block barrier is needed (compiler fence only).
// ---------------------------------------------------------------------------
__global__ __launch_bounds__(128) void attn_kernel(const f16* __restrict__ Q,
                                                   const f16* __restrict__ K,
                                                   const f16* __restrict__ Vt,
                                                   float* __restrict__ out) {
    __shared__ f16 Pl[4][16 * 72];   // per-wave 16x64 P tile, stride 72 halves

    const int tid  = threadIdx.x;
    const int wv   = tid >> 5;
    const int lane = tid & 31;
    const int lo   = lane & 15;
    const int hi   = lane >> 4;
    const int bh   = blockIdx.y;               // b*NH + h
    const int qrow0 = blockIdx.x * 64 + wv * 16;

    // Q A-fragments, resident for the whole K loop (Q pre-scaled by 1/sqrt(d))
    v16h qa[3];
    {
        const f16* qrow = Q + ((size_t)bh * SEQL + qrow0 + lo) * HD;
#pragma unroll
        for (int c = 0; c < 3; ++c) {
            const f16* p = qrow + c * 32 + 8 * hi;
            qa[c] = join16(*(const v8h*)p, *(const v8h*)(p + 16));
        }
    }

    const f16* Kb = K  + (size_t)bh * SEQL * HD;
    const f16* Vb = Vt + (size_t)bh * HD * SEQL;

    v8f o[6] = {};
    float mrow[8], lrow[8];
#pragma unroll
    for (int v = 0; v < 8; ++v) { mrow[v] = -1e30f; lrow[v] = 0.0f; }

    for (int kt = 0; kt < SEQL; kt += 64) {
        // prefetch next K/V tiles into L2/L0 (global_prefetch_b8)
        if (kt + 64 < SEQL) {
            __builtin_prefetch(Kb + (size_t)(kt + 64 + lane) * HD, 0, 0);
            __builtin_prefetch(Kb + (size_t)(kt + 96 + lane) * HD, 0, 0);
            __builtin_prefetch(Vb + (size_t)(lane * 3) * SEQL + kt + 64, 0, 0);
        }

        // ---- scores S = Q @ K^T for 64 k-columns (four 16-wide fragments) ----
        v8f s[4] = {};
#pragma unroll
        for (int c = 0; c < 3; ++c) {
#pragma unroll
            for (int nt = 0; nt < 4; ++nt) {
                // B fragment: k-row = kt + nt*16 + lo, d = c*32 + 8*hi .. +8, c*32 + 16 + 8*hi .. +8
                const f16* p = Kb + (size_t)(kt + nt * 16 + lo) * HD + c * 32 + 8 * hi;
                v16h b = join16(*(const v8h*)p, *(const v8h*)(p + 16));
                s[nt] = __builtin_amdgcn_wmma_f32_16x16x32_f16(false, qa[c], false, b,
                                                               (short)0, s[nt], false, false);
            }
        }

        // ---- online softmax (row stats reduced across 16 lanes per half) ----
        float alpha[8];
#pragma unroll
        for (int v = 0; v < 8; ++v) {
            float t = fmaxf(fmaxf(s[0][v], s[1][v]), fmaxf(s[2][v], s[3][v]));
            t = fmaxf(t, __shfl_xor(t, 1, 32));
            t = fmaxf(t, __shfl_xor(t, 2, 32));
            t = fmaxf(t, __shfl_xor(t, 4, 32));
            t = fmaxf(t, __shfl_xor(t, 8, 32));
            float mnew = fmaxf(mrow[v], t);
            alpha[v] = __expf(mrow[v] - mnew);
            mrow[v] = mnew;
            float rs = 0.0f;
#pragma unroll
            for (int nt = 0; nt < 4; ++nt) {
                float p = __expf(s[nt][v] - mnew);
                s[nt][v] = p;
                rs += p;
            }
            rs += __shfl_xor(rs, 1, 32);
            rs += __shfl_xor(rs, 2, 32);
            rs += __shfl_xor(rs, 4, 32);
            rs += __shfl_xor(rs, 8, 32);
            lrow[v] = lrow[v] * alpha[v] + rs;
        }
#pragma unroll
        for (int dt = 0; dt < 6; ++dt)
#pragma unroll
            for (int v = 0; v < 8; ++v) o[dt][v] *= alpha[v];

        // ---- P (C layout) -> wave-private LDS -> two A-layout fragments ----
        f16* P = Pl[wv];
#pragma unroll
        for (int v = 0; v < 8; ++v) {
            int row = (v + 8 * hi) * 72;
#pragma unroll
            for (int nt = 0; nt < 4; ++nt)
                P[row + nt * 16 + lo] = (f16)s[nt][v];
        }
        asm volatile("" ::: "memory");   // same-wave DS ops are in-order; fence compiler only
        v16h pa[2];
#pragma unroll
        for (int j = 0; j < 2; ++j) {
            const f16* pp = P + lo * 72 + j * 32 + 8 * hi;
            pa[j] = join16(*(const v8h*)pp, *(const v8h*)(pp + 16));
        }
        asm volatile("" ::: "memory");

        // ---- O += P @ V  (Vt layout: column d contiguous over k) ----
#pragma unroll
        for (int dt = 0; dt < 6; ++dt) {
#pragma unroll
            for (int j = 0; j < 2; ++j) {
                const f16* vp = Vb + (size_t)(dt * 16 + lo) * SEQL + kt + j * 32 + 8 * hi;
                v16h b = join16(*(const v8h*)vp, *(const v8h*)(vp + 16));
                o[dt] = __builtin_amdgcn_wmma_f32_16x16x32_f16(false, pa[j], false, b,
                                                               (short)0, o[dt], false, false);
            }
        }
    }

    // ---- epilogue: normalize by l, store fp32 [B, L, 768] ----
    const int b = bh >> 3;
    const int h = bh & 7;
#pragma unroll
    for (int v = 0; v < 8; ++v) {
        float inv = 1.0f / lrow[v];
        int q = qrow0 + v + 8 * hi;
        float* orow = out + ((size_t)b * SEQL + q) * DM + h * HD;
#pragma unroll
        for (int dt = 0; dt < 6; ++dt) orow[dt * 16 + lo] = o[dt][v] * inv;
    }
}

extern "C" void kernel_launch(void* const* d_in, const int* in_sizes, int n_in,
                              void* d_out, int out_size, void* d_ws, size_t ws_size,
                              hipStream_t stream) {
    (void)in_sizes; (void)n_in; (void)out_size; (void)ws_size;
    const float* hs = (const float*)d_in[0];
    const float* kv = (const float*)d_in[1];
    const float* Wq = (const float*)d_in[2];
    const float* Wk = (const float*)d_in[3];
    const float* Wv = (const float*)d_in[4];
    float* out = (float*)d_out;

    f16* ws = (f16*)d_ws;
    const size_t per = (size_t)BATCH * NH * SEQL * HD;   // 6.29M halves each
    f16* Qb  = ws;
    f16* Kb  = ws + per;
    f16* Vtb = ws + 2 * per;

    dim3 blk(128, 1, 1);
    dim3 pgrid(DM / 64, (BATCH * SEQL) / 64, 1);         // 12 x 128 tiles
    const float qscale = 1.0f / sqrtf((float)HD);

    proj_kernel<<<pgrid, blk, 0, stream>>>(hs, Wq, Qb, 0, qscale);
    proj_kernel<<<pgrid, blk, 0, stream>>>(kv, Wk, Kb, 1, 1.0f);
    proj_kernel<<<pgrid, blk, 0, stream>>>(kv, Wv, Vtb, 2, 1.0f);

    dim3 agrid(SEQL / 64, BATCH * NH, 1);                // 64 x 16 blocks
    attn_kernel<<<agrid, blk, 0, stream>>>(Qb, Kb, Vtb, out);
}
